// HybridQuantumRegressionModel_65481071400626
// MI455X (gfx1250) — matrix operand order repaired and checked by reading, not compiled, not run on hardware
//
#include <hip/hip_runtime.h>

typedef _Float16 v16h __attribute__((ext_vector_type(16)));
typedef float    v8f  __attribute__((ext_vector_type(8)));
typedef float    v2f  __attribute__((ext_vector_type(2)));

// ---------------- complex helpers ----------------
__device__ __forceinline__ float2 cmul(float2 a, float2 b) {
  return make_float2(a.x * b.x - a.y * b.y, a.x * b.y + a.y * b.x);
}
__device__ __forceinline__ float2 cadd(float2 a, float2 b) {
  return make_float2(a.x + b.x, a.y + b.y);
}

// ---------------- Kernel 1: conv1 (1->8, 3x3 SAME) + ReLU + maxpool2 -> f16 h1[b][8][32][32]
__global__ void conv1_relu_pool_kernel(const float* __restrict__ x,
                                       const float* __restrict__ w1,
                                       const float* __restrict__ b1,
                                       _Float16* __restrict__ h1) {
  __shared__ float sw[72];
  __shared__ float sb[8];
  const int tid = threadIdx.x;
  if (tid < 72) sw[tid] = w1[tid];
  if (tid < 8)  sb[tid] = b1[tid];
  __syncthreads();

  const int bid = blockIdx.x;                  // 0..4095 (4 blocks per image)
  const int b = bid >> 2, q = bid & 3;
  const int oy = ((q >> 1) << 4) + (tid >> 4); // pooled row 0..31
  const int ox = ((q & 1) << 4) + (tid & 15);  // pooled col 0..31
  const float* xb = x + (size_t)b * 4096;

  float patch[4][4];
#pragma unroll
  for (int dy = 0; dy < 4; ++dy) {
    int iy = 2 * oy - 1 + dy;
#pragma unroll
    for (int dx = 0; dx < 4; ++dx) {
      int ix = 2 * ox - 1 + dx;
      patch[dy][dx] = ((unsigned)iy < 64u && (unsigned)ix < 64u) ? xb[iy * 64 + ix] : 0.f;
    }
  }
#pragma unroll
  for (int c = 0; c < 8; ++c) {
    float mx = -1e30f;
#pragma unroll
    for (int py = 0; py < 2; ++py)
#pragma unroll
      for (int px = 0; px < 2; ++px) {
        float s = sb[c];
#pragma unroll
        for (int ky = 0; ky < 3; ++ky)
#pragma unroll
          for (int kx = 0; kx < 3; ++kx)
            s = fmaf(patch[py + ky][px + kx], sw[c * 9 + ky * 3 + kx], s);
        mx = fmaxf(mx, s);
      }
    h1[(((size_t)b * 8 + c) << 10) + (oy << 5) + ox] = (_Float16)fmaxf(mx, 0.f);
  }
}

// ---------------- Kernel 2: pack conv2 weights into WMMA f16 A-matrix layout (3 K-chunks)
// A element (m,k): lane = m + 16*((k>>3)&1), half = (k&7) + (k>>4)*8
__global__ void pack_w2_kernel(const float* __restrict__ w2, _Float16* __restrict__ pack) {
  int t = blockIdx.x * 256 + threadIdx.x;
  if (t >= 1536) return;
  int h = t & 15, lane = (t >> 4) & 31, c = t >> 9;
  int m = lane & 15;
  int k = (h & 7) + ((h >> 3) << 4) + ((lane >> 4) << 3);
  int kk = c * 32 + k;
  float v = 0.f;
  if (kk < 72) {
    int ci = kk / 9, rem = kk - ci * 9;
    int ky = rem / 3, kx = rem - ky * 3;
    v = w2[((m * 8 + ci) * 3 + ky) * 3 + kx]; // OIHW (16,8,3,3)
  }
  pack[t] = (_Float16)v;
}

// ---------------- Kernel 3: conv2 implicit GEMM via WMMA f16 + ReLU + maxpool + global-avg-pool
// Padded LDS image: 8 planes of 34 rows x 36 cols (zero border), plane 8 = all zeros.
// Gather address = base(py,px) + koff(kk); no bounds checks, koff is compile-time.
// N-dim of each WMMA = 16 POOLED positions; the 4 pool sub-positions are separate
// accumulations folded with a per-lane running max => no cross-lane pooling traffic.
#define PLANE 1224                 // 34 * 36
#define ZOFF  (8 * PLANE)          // all-zero plane (covers OOB K 72..95)

__host__ __device__ constexpr int koff(int kk) {
  return (kk < 72) ? ((kk / 9) * PLANE + ((kk % 9) / 3) * 36 + ((kk % 9) % 3))
                   : ZOFF;
}

__global__ __launch_bounds__(128, 1)
void conv2_wmma_kernel(const _Float16* __restrict__ h1,
                       const _Float16* __restrict__ w2pack,
                       const float* __restrict__ b2,
                       float* __restrict__ pooled) {
  __shared__ __align__(16) _Float16 sh[9 * PLANE];   // 22032 B
  __shared__ float sfeat[16];
  const int tid = threadIdx.x, b = blockIdx.x;
  if (tid < 16) sfeat[tid] = 0.f;

  // zero borders + zero plane (whole region)
  {
    unsigned int* z = reinterpret_cast<unsigned int*>(sh);
    for (int i = tid; i < (9 * PLANE) / 2; i += 128) z[i] = 0u;
  }
  __syncthreads();
  // interior fill: 8 contiguous halves per float4 read, 4 aligned u32 LDS stores
  {
    const float4* src = reinterpret_cast<const float4*>(h1 + (size_t)b * 8192);
    for (int i = tid; i < 1024; i += 128) {
      float4 v = src[i];
      int ci = i >> 7;              // 128 float4 per 32x32 plane
      int rr = (i >> 2) & 31;       // 4 float4 per row
      int c8 = (i & 3) << 3;        // col start, multiple of 8
      int base = ci * PLANE + (rr + 1) * 36 + (c8 + 2);   // even -> 4B aligned
      unsigned int* d = reinterpret_cast<unsigned int*>(sh + base);
      const unsigned int* s = reinterpret_cast<const unsigned int*>(&v);
      d[0] = s[0]; d[1] = s[1]; d[2] = s[2]; d[3] = s[3];
    }
  }
  __syncthreads();

  const int wv = tid >> 5, lane = tid & 31;
  const int n = lane & 15;
  const bool hi = (lane >> 4) != 0;

  v16h A[3];
#pragma unroll
  for (int c = 0; c < 3; ++c)
    A[c] = *reinterpret_cast<const v16h*>(w2pack + (c * 32 + lane) * 16);
  float bias[8];
#pragma unroll
  for (int r = 0; r < 8; ++r) bias[r] = b2[r + (hi ? 8 : 0)];

  float acc[8];
#pragma unroll
  for (int r = 0; r < 8; ++r) acc[r] = 0.f;

  for (int mt = wv; mt < 16; mt += 4) {    // 16 macro-tiles of 4x4 POOLED positions
    const int qy = ((mt >> 2) << 2) + (n >> 2);   // pooled coords 0..15
    const int qx = ((mt & 3) << 2) + (n & 3);

    float m8[8];
#pragma unroll
    for (int sub = 0; sub < 4; ++sub) {    // 2x2 pool window sub-positions
      const int py = 2 * qy + (sub >> 1);
      const int px = 2 * qx + (sub & 1);
      const int base = py * 36 + px + 1;   // addr = base + koff(kk), always in-range

      v8f cAcc = {};
#pragma unroll
      for (int c = 0; c < 3; ++c) {
        v16h B;                            // B element (k,n): lane = n + 16*((k>>3)&1), half = (k&7) + (k>>4)*8
#pragma unroll
        for (int h = 0; h < 16; ++h) {
          int off = hi ? koff(c * 32 + 8 + (h & 7) + ((h >> 3) << 4)) : koff(c * 32 + (h & 7) + ((h >> 3) << 4));  // cndmask of immediates
          B[h] = sh[base + off];
        }
        cAcc = __builtin_amdgcn_wmma_f32_16x16x32_f16(false, A[c], false, B, (short)0, cAcc, false, false);
      }
#pragma unroll
      for (int r = 0; r < 8; ++r)
        m8[r] = (sub == 0) ? cAcc[r] : fmaxf(m8[r], cAcc[r]);  // register-local maxpool
    }
#pragma unroll
    for (int r = 0; r < 8; ++r)
      acc[r] += fmaxf(m8[r] + bias[r], 0.f);                    // relu(max(conv)+b)
  }

  // reduce avg-pool partial sums across the 16 pooled-position lanes (per half-wave)
#pragma unroll
  for (int r = 0; r < 8; ++r) {
    float s = acc[r];
    s += __shfl_xor(s, 1, 32);
    s += __shfl_xor(s, 2, 32);
    s += __shfl_xor(s, 4, 32);
    s += __shfl_xor(s, 8, 32);
    if (n == 0) atomicAdd(&sfeat[r + (hi ? 8 : 0)], s);         // ds_add_f32, 16 per wave
  }
  __syncthreads();
  if (tid < 16) pooled[b * 16 + tid] = sfeat[tid] * (1.f / 256.f);
}

// ---------------- MT19937 (numpy RandomState legacy) ----------------
__device__ unsigned int mt_next(unsigned int* mt, int& mti) {
  if (mti >= 624) {
    for (int i = 0; i < 624; ++i) {
      unsigned int y = (mt[i] & 0x80000000u) | (mt[(i + 1) % 624] & 0x7fffffffu);
      mt[i] = mt[(i + 397) % 624] ^ (y >> 1) ^ ((y & 1u) ? 0x9908b0dfu : 0u);
    }
    mti = 0;
  }
  unsigned int y = mt[mti++];
  y ^= y >> 11;
  y ^= (y << 7) & 0x9d2c5680u;
  y ^= (y << 15) & 0xefc60000u;
  y ^= y >> 18;
  return y;
}

// ---------------- Kernel 4: rebuild RAND_OPS spec + compose fixed 16x16 unitary U
__global__ void build_unitary_kernel(const float* __restrict__ rl,
                                     const float* __restrict__ qp,
                                     float2* __restrict__ U) {
  __shared__ unsigned int mt[624];
  __shared__ int gg[57], ga[57], gb[57];
  __shared__ float2 st[256];
  const int tid = threadIdx.x;
  if (tid == 0) {
    mt[0] = 42u;
    for (int i = 1; i < 624; ++i)
      mt[i] = 1812433253u * (mt[i - 1] ^ (mt[i - 1] >> 30)) + (unsigned)i;
    int mti = 624;
    for (int i = 0; i < 50; ++i) {
      (void)mt_next(mt, mti);                    // randint(4): uint64 draw, high word
      int g = (int)(mt_next(mt, mti) & 3u);      // low word & mask(3), no rejection
      gg[i] = g; gb[i] = 0;
      if (g < 3) {
        (void)mt_next(mt, mti);                  // randint(N_WIRES)
        ga[i] = (int)(mt_next(mt, mti) & 3u);
      } else {                                   // choice(4,2,replace=False) = permutation(4)[:2]
        int arr[4] = {0, 1, 2, 3};
        for (int ii = 3; ii >= 1; --ii) {
          unsigned mask = (unsigned)ii;
          mask |= mask >> 1; mask |= mask >> 2; mask |= mask >> 4;
          mask |= mask >> 8; mask |= mask >> 16;
          unsigned j;
          do { j = mt_next(mt, mti) & mask; } while (j > (unsigned)ii);
          int tmp = arr[ii]; arr[ii] = arr[j]; arr[j] = tmp;
        }
        ga[i] = arr[0]; gb[i] = arr[1];
      }
    }
    // fixed gates: rx0@0, ry0@1, rz0@3, crx0@(0,2), H@3, SX@2, CNOT@(3,0)
    gg[50] = 0; ga[50] = 0; gb[50] = 0;
    gg[51] = 1; ga[51] = 1; gb[51] = 0;
    gg[52] = 2; ga[52] = 3; gb[52] = 0;
    gg[53] = 3; ga[53] = 0; gb[53] = 2;
    gg[54] = 4; ga[54] = 3; gb[54] = 0;
    gg[55] = 5; ga[55] = 2; gb[55] = 0;
    gg[56] = 6; ga[56] = 3; gb[56] = 0;
  }
  __syncthreads();
  if (tid < 16) {
    float2* v = &st[tid * 16];                   // column tid of U: evolve basis state
    for (int i = 0; i < 16; ++i) v[i] = make_float2(i == tid ? 1.f : 0.f, 0.f);
    for (int op = 0; op < 57; ++op) {
      int g = gg[op], a = ga[op];
      float t = (op < 50) ? rl[op] : ((op < 54) ? qp[op - 50] : 0.f);
      float ch = cosf(0.5f * t), sn = sinf(0.5f * t);
      float2 u00, u01, u10, u11;
      bool ctrl = false; int tw = a;
      switch (g) {
        case 0: u00=make_float2(ch,0); u01=make_float2(0,-sn); u10=make_float2(0,-sn); u11=make_float2(ch,0); break;
        case 1: u00=make_float2(ch,0); u01=make_float2(-sn,0); u10=make_float2(sn,0);  u11=make_float2(ch,0); break;
        case 2: u00=make_float2(ch,-sn); u01=make_float2(0,0); u10=make_float2(0,0);   u11=make_float2(ch,sn); break;
        case 3: ctrl=true; tw=gb[op];
                u00=make_float2(ch,0); u01=make_float2(0,-sn); u10=make_float2(0,-sn); u11=make_float2(ch,0); break;
        case 4: { float r_=0.7071067811865476f;
                u00=make_float2(r_,0); u01=make_float2(r_,0); u10=make_float2(r_,0); u11=make_float2(-r_,0); } break;
        case 5: u00=make_float2(0.5f,0.5f); u01=make_float2(0.5f,-0.5f);
                u10=make_float2(0.5f,-0.5f); u11=make_float2(0.5f,0.5f); break;
        default: ctrl=true; tw=gb[op];          // CNOT: controlled-X
                u00=make_float2(0,0); u01=make_float2(1.f,0); u10=make_float2(1.f,0); u11=make_float2(0,0); break;
      }
      int cb = ctrl ? (1 << (3 - a)) : 0;       // wire 0 = MSB of amplitude index
      int tb = 1 << (3 - tw);
      for (int i = 0; i < 16; ++i) {
        if (i & tb) continue;
        if (ctrl && !(i & cb)) continue;
        int i1 = i | tb;
        float2 xv = v[i], yv = v[i1];
        v[i]  = cadd(cmul(u00, xv), cmul(u01, yv));
        v[i1] = cadd(cmul(u10, xv), cmul(u11, yv));
      }
    }
    for (int i = 0; i < 16; ++i) U[i * 16 + tid] = v[i];  // row-major U[m][k]
  }
}

// ---------------- Kernel 5: batched quantum evolve (f32 WMMA complex GEMM) + measure + head + BN
__global__ __launch_bounds__(128, 1)
void quantum_head_kernel(const float* __restrict__ pooled,
                         const float2* __restrict__ U,
                         const float* __restrict__ hw,
                         const float* __restrict__ hb,
                         const float* __restrict__ bng,
                         const float* __restrict__ bnb,
                         const float* __restrict__ bnm,
                         const float* __restrict__ bnv,
                         float* __restrict__ out) {
  const int wave = (blockIdx.x * blockDim.x + threadIdx.x) >> 5;  // 0..63, 16 batch elems each
  const int lane = threadIdx.x & 31;
  const int n = lane & 15, hi = lane >> 4;
  const int bidx = wave * 16 + n;
  const float* p = pooled + bidx * 16;

  // encoder product state: per-wire 2-vector = RY(p12+w) RX(p8+w) RZ(p4+w) RY(pw) |0>
  float2 V[4][2];
#pragma unroll
  for (int w = 0; w < 4; ++w) {
    float c0 = cosf(0.5f * p[w]),      s0 = sinf(0.5f * p[w]);
    float cz = cosf(0.5f * p[4 + w]),  sz = sinf(0.5f * p[4 + w]);
    float cx = cosf(0.5f * p[8 + w]),  sx = sinf(0.5f * p[8 + w]);
    float cy = cosf(0.5f * p[12 + w]), sy = sinf(0.5f * p[12 + w]);
    float2 v0 = make_float2(c0 * cz, -c0 * sz);
    float2 v1 = make_float2(s0 * cz,  s0 * sz);
    float2 n0 = make_float2(cx * v0.x + sx * v1.y,  cx * v0.y - sx * v1.x);  // c*v0 - i*s*v1
    float2 n1 = make_float2(sx * v0.y + cx * v1.x, -sx * v0.x + cx * v1.y);  // -i*s*v0 + c*v1
    V[w][0] = make_float2(cy * n0.x - sy * n1.x, cy * n0.y - sy * n1.y);
    V[w][1] = make_float2(sy * n0.x + cy * n1.x, sy * n0.y + cy * n1.y);
  }
  float2 pa[4], pb[4];
#pragma unroll
  for (int a2 = 0; a2 < 4; ++a2) {
    pa[a2] = cmul(V[0][a2 >> 1], V[1][a2 & 1]);
    pb[a2] = cmul(V[2][a2 >> 1], V[3][a2 & 1]);
  }

  // PSI'[16 x 16batch] = U[16x16] x PSI[16 x 16batch], complex, K split into 4 chunks of 4
  v2f are[4], aimn[4], bre[4], bim[4];
#pragma unroll
  for (int kc = 0; kc < 4; ++kc) {
    int k0 = kc * 4 + 2 * hi;
    float2 A0 = cmul(pa[k0 >> 2], pb[k0 & 3]);
    float2 A1 = cmul(pa[(k0 + 1) >> 2], pb[(k0 + 1) & 3]);
    bre[kc][0] = A0.x; bre[kc][1] = A1.x;
    bim[kc][0] = A0.y; bim[kc][1] = A1.y;
    float2 u0 = U[n * 16 + k0];
    float2 u1 = U[n * 16 + k0 + 1];
    are[kc][0]  = u0.x;  are[kc][1]  = u1.x;
    aimn[kc][0] = -u0.y; aimn[kc][1] = -u1.y;   // pre-negated Uim (f32 WMMA has no A/B neg modifier)
  }
  v8f dre = {}, dimg = {};
#pragma unroll
  for (int kc = 0; kc < 4; ++kc) {
    v2f aim = -aimn[kc];                         // +Uim for the imaginary-part products
    dre  = __builtin_amdgcn_wmma_f32_16x16x4_f32(false, are[kc],  false, bre[kc], (short)0, dre,  false, false);
    dre  = __builtin_amdgcn_wmma_f32_16x16x4_f32(false, aimn[kc], false, bim[kc], (short)0, dre,  false, false);
    dimg = __builtin_amdgcn_wmma_f32_16x16x4_f32(false, are[kc],  false, bim[kc], (short)0, dimg, false, false);
    dimg = __builtin_amdgcn_wmma_f32_16x16x4_f32(false, aim,      false, bre[kc], (short)0, dimg, false, false);
  }

  // probs -> <Z_w> features; D layout: lane holds amplitudes m = r + 8*hi for batch col n
  float f0 = 0, f1 = 0, f2 = 0, f3 = 0;
#pragma unroll
  for (int r = 0; r < 8; ++r) {
    int m = r + 8 * hi;
    float pr = dre[r] * dre[r] + dimg[r] * dimg[r];
    f0 += ((m >> 3) & 1) ? -pr : pr;
    f1 += ((m >> 2) & 1) ? -pr : pr;
    f2 += ((m >> 1) & 1) ? -pr : pr;
    f3 += (m & 1) ? -pr : pr;
  }
  f0 += __shfl_xor(f0, 16, 32);
  f1 += __shfl_xor(f1, 16, 32);
  f2 += __shfl_xor(f2, 16, 32);
  f3 += __shfl_xor(f3, 16, 32);
  if (hi == 0) {
    float o = f0 * hw[0] + f1 * hw[1] + f2 * hw[2] + f3 * hw[3] + hb[0];
    o = bng[0] * (o - bnm[0]) * rsqrtf(bnv[0] + 1e-5f) + bnb[0];
    out[bidx] = o;
  }
}

// ---------------- host launcher ----------------
extern "C" void kernel_launch(void* const* d_in, const int* in_sizes, int n_in,
                              void* d_out, int out_size, void* d_ws, size_t ws_size,
                              hipStream_t stream) {
  (void)in_sizes; (void)n_in; (void)out_size; (void)ws_size;
  const float* x   = (const float*)d_in[0];
  const float* w1  = (const float*)d_in[1];
  const float* b1  = (const float*)d_in[2];
  const float* w2  = (const float*)d_in[3];
  const float* b2  = (const float*)d_in[4];
  const float* rl  = (const float*)d_in[5];
  const float* qp  = (const float*)d_in[6];
  const float* hw  = (const float*)d_in[7];
  const float* hb  = (const float*)d_in[8];
  const float* bng = (const float*)d_in[9];
  const float* bnb = (const float*)d_in[10];
  const float* bnm = (const float*)d_in[11];
  const float* bnv = (const float*)d_in[12];
  float* out = (float*)d_out;

  char* ws = (char*)d_ws;
  _Float16* h1     = (_Float16*)(ws);                                  // 16 MiB: 1024x8x32x32 f16
  float*    pooled = (float*)   (ws + (size_t)(16u << 20));            // 64 KiB: 1024x16 f32
  _Float16* w2p    = (_Float16*)(ws + (size_t)(16u << 20) + 65536);    // 3 KiB packed weights
  float2*   U      = (float2*)  (ws + (size_t)(16u << 20) + 65536 + 4096); // 2 KiB unitary

  pack_w2_kernel<<<6, 256, 0, stream>>>(w2, w2p);
  build_unitary_kernel<<<1, 32, 0, stream>>>(rl, qp, U);
  conv1_relu_pool_kernel<<<4096, 256, 0, stream>>>(x, w1, b1, h1);
  conv2_wmma_kernel<<<1024, 128, 0, stream>>>(h1, w2p, b2, pooled);
  quantum_head_kernel<<<16, 128, 0, stream>>>(pooled, U, hw, hb, bng, bnb, bnm, bnv, out);
}
